// SemanticModel_88510686036290
// MI455X (gfx1250) — matrix operand order repaired and checked by reading, not compiled, not run on hardware
//
#include <hip/hip_runtime.h>
#include <cstdint>
#include <cstddef>

// MI455X / gfx1250, wave32. Sizes fixed by the reference problem.
#define NN    4096
#define NEMB  2048
#define NHID  1024
#define MM    8
#define DD    512

typedef __attribute__((ext_vector_type(16))) __bf16 v16bf;
typedef __attribute__((ext_vector_type(8)))  float  v8f;

// ---------------- small helpers ----------------
__device__ inline void load4f(const float* p, float f[4]) {
  const float4 v = *(const float4*)p;
  f[0] = v.x; f[1] = v.y; f[2] = v.z; f[3] = v.w;
}
__device__ inline void pack4_store(__bf16* dst, const float f[4]) {
  union { __bf16 h[4]; uint64_t u; } pk;
  pk.h[0] = (__bf16)f[0]; pk.h[1] = (__bf16)f[1];
  pk.h[2] = (__bf16)f[2]; pk.h[3] = (__bf16)f[3];
  *(uint64_t*)dst = pk.u;
}

// CDNA5 async copy: 16 bytes global -> LDS, per-lane LDS destination.
// GVS mode: saddr = 64-bit SGPR base, vaddr = 32-bit per-lane byte offset,
// vdst = 32-bit LDS byte address. Tracked by ASYNCcnt.
__device__ inline void async_ld_b128(uint32_t lds_addr, uint32_t goff, const void* base) {
  asm volatile("global_load_async_to_lds_b128 %0, %1, %2"
               :: "v"(lds_addr), "v"(goff), "s"(base)
               : "memory");
}

// ---------------- one-time conversions ----------------
// f32 -> bf16 (row-major copy), 4 elements/thread
__global__ __launch_bounds__(256) void cvt_bf16_kernel(
    const float* __restrict__ in, __bf16* __restrict__ out) {
  const int i = blockIdx.x * 256 + threadIdx.x;
  float f[4];
  load4f(in + (size_t)i * 4, f);
  pack4_store(out + (size_t)i * 4, f);
}

// W[K,1024] f32 -> Wt[1024,K] bf16 (transpose so K is contiguous).
// Reads coalesced across lanes (consecutive n); 32B writes per thread.
__global__ __launch_bounds__(256) void transpose_cvt_kernel(
    const float* __restrict__ W, __bf16* __restrict__ Wt, int K) {
  const int g  = blockIdx.x * 256 + threadIdx.x;
  const int n  = g & (NHID - 1);
  const int k0 = (g >> 10) << 4;
  union { __bf16 h[16]; uint4 u[2]; } pk;
  #pragma unroll
  for (int j = 0; j < 16; ++j)
    pk.h[j] = (__bf16)W[(size_t)(k0 + j) * NHID + n];
  uint4* dst = (uint4*)(Wt + (size_t)n * K + k0);
  dst[0] = pk.u[0];
  dst[1] = pk.u[1];
}

// ---------------- async tile staging into WMMA fragment layout ----------------
// A (16x32 bf16, MxK): lane = (m&15) + 16*((k>>3)&1), elem i = (k>>4)*8 + (k&7)
// B (32x16 bf16, KxN): lane = (n&15) + 16*((k>>3)&1), elem i = (k>>4)*8 + (k&7)
// Both operands are K-contiguous in global (A row-major, Bt transposed), so each
// 8-bf16 K-segment is one contiguous 16B global read -> one contiguous 16B LDS
// write. 4 async instructions per wave per 32-deep K chunk.
__device__ inline void stage_tiles(const __bf16* A, int lda, const __bf16* Bt, int K,
                                   int rowBase, int colBase, int k0, int tid,
                                   v16bf (*aF)[32], v16bf (*bF)[32])
{
  #pragma unroll
  for (int s = 0; s < 2; ++s) {
    const int seg  = tid + (s << 8);          // 0..511
    const int m    = seg >> 2;                // 0..127
    const int kseg = (seg & 3) << 3;          // 0,8,16,24
    const uint32_t goff = (uint32_t)(((rowBase + m) * lda + (k0 + kseg)) << 1);
    const int l = (m & 15) + (((kseg >> 3) & 1) << 4);
    const uint32_t lds = (uint32_t)(uintptr_t)&aF[m >> 4][l]
                       + (uint32_t)(((kseg >> 4) & 1) << 4);
    async_ld_b128(lds, goff, A);
  }
  #pragma unroll
  for (int s = 0; s < 2; ++s) {
    const int seg  = tid + (s << 8);
    const int n    = seg >> 2;
    const int kseg = (seg & 3) << 3;
    const uint32_t goff = (uint32_t)(((colBase + n) * K + (k0 + kseg)) << 1);
    const int l = (n & 15) + (((kseg >> 3) & 1) << 4);
    const uint32_t lds = (uint32_t)(uintptr_t)&bF[n >> 4][l]
                       + (uint32_t)(((kseg >> 4) & 1) << 4);
    async_ld_b128(lds, goff, Bt);
  }
}

// ---------------- fused GEMM + bias + ELU, bf16 WMMA, double-buffered ----------
__global__ __launch_bounds__(256) void gemm_bias_elu_wmma(
    const __bf16* __restrict__ A,   int lda,    // [rows, K] row-major bf16
    const __bf16* __restrict__ Bt,              // [NHID, K] (transposed) bf16
    const float*  __restrict__ bias,
    __bf16*       __restrict__ out, int ldo,
    int K)
{
  __shared__ v16bf aFrag[2][8][32];   // 16 KB : [buf][m_tile][lane]
  __shared__ v16bf bFrag[2][8][32];   // 16 KB : [buf][n_tile][lane]

  const int tid  = threadIdx.x;
  const int lane = tid & 31;
  const int wave = tid >> 5;
  const int wm   = wave & 3;            // 4 waves over M: 32 rows each
  const int wn   = wave >> 2;           // 2 waves over N: 64 cols each
  const int rowBase = blockIdx.y * 128;
  const int colBase = blockIdx.x * 128;

  v8f acc[8] = {};                      // [m_tile(2)][n_tile(4)] f32 accumulators

  const int kChunks = K >> 5;
  stage_tiles(A, lda, Bt, K, rowBase, colBase, 0, tid, aFrag[0], bFrag[0]);

  for (int kc = 0; kc < kChunks; ++kc) {
    const int buf = kc & 1;
    if (kc + 1 < kChunks) {
      // prefetch next chunk into the other buffer (its readers finished at the
      // trailing barrier of iteration kc-1)
      stage_tiles(A, lda, Bt, K, rowBase, colBase, (kc + 1) << 5, tid,
                  aFrag[buf ^ 1], bFrag[buf ^ 1]);
      // async loads complete in order: <=4 outstanding => chunk kc has landed
      asm volatile("s_wait_asynccnt 0x4" ::: "memory");
    } else {
      asm volatile("s_wait_asynccnt 0x0" ::: "memory");
    }
    __syncthreads();

    const v16bf a0 = aFrag[buf][wm * 2 + 0][lane];
    const v16bf a1 = aFrag[buf][wm * 2 + 1][lane];
    #pragma unroll
    for (int j = 0; j < 4; ++j) {
      const v16bf b = bFrag[buf][wn * 4 + j][lane];
      acc[j]     = __builtin_amdgcn_wmma_f32_16x16x32_bf16(
          false, a0, false, b, (short)0, acc[j],     false, false);
      acc[4 + j] = __builtin_amdgcn_wmma_f32_16x16x32_bf16(
          false, a1, false, b, (short)0, acc[4 + j], false, false);
    }
    __syncthreads();
  }

  // epilogue: bias + ELU, bf16 store (C/D layout: row=(lane>>4)*8+v, col=lane&15)
  #pragma unroll
  for (int i = 0; i < 2; ++i) {
    #pragma unroll
    for (int j = 0; j < 4; ++j) {
      const v8f c = acc[i * 4 + j];
      const int col = colBase + wn * 64 + j * 16 + (lane & 15);
      const float bv = bias[col];
      #pragma unroll
      for (int v = 0; v < 8; ++v) {
        const int row = rowBase + wm * 32 + i * 16 + ((lane >> 4) << 3) + v;
        float x = c[v] + bv;
        x = x > 0.0f ? x : (__expf(x) - 1.0f);
        out[(size_t)row * ldo + col] = (__bf16)x;
      }
    }
  }
}

// ---------------- B = h2 @ W3 + b3  ([4096,3], wave-per-row) ----------------
__global__ __launch_bounds__(256) void coef_kernel(
    const __bf16* __restrict__ h, const float* __restrict__ W3,
    const float* __restrict__ b3, float* __restrict__ Bc)
{
  const int row  = blockIdx.x * 8 + (threadIdx.x >> 5);
  const int lane = threadIdx.x & 31;
  float s0 = 0.f, s1 = 0.f, s2 = 0.f;
  const __bf16* hr = h + (size_t)row * NHID;
  for (int k = lane; k < NHID; k += 32) {
    const float v = (float)hr[k];
    s0 += v * W3[k * 3 + 0];
    s1 += v * W3[k * 3 + 1];
    s2 += v * W3[k * 3 + 2];
  }
  #pragma unroll
  for (int off = 16; off > 0; off >>= 1) {
    s0 += __shfl_xor(s0, off, 32);
    s1 += __shfl_xor(s1, off, 32);
    s2 += __shfl_xor(s2, off, 32);
  }
  if (lane == 0) {
    Bc[row * 3 + 0] = s0 + b3[0];
    Bc[row * 3 + 1] = s1 + b3[1];
    Bc[row * 3 + 2] = s2 + b3[2];
  }
}

// ---------------- y[n,m,d] = sum_c B[n,c] * X[n,m,c,d]  (bandwidth bound) ----
__global__ __launch_bounds__(256) void apply_kernel(
    const float* __restrict__ Bc, const float* __restrict__ X,
    float* __restrict__ y)
{
  const int idx = blockIdx.x * 256 + threadIdx.x;  // float4 index, < 4194304
  const int d4  = idx & (DD / 4 - 1);              // 0..127
  const int nm  = idx >> 7;                        // n*MM + m
  const int n   = nm >> 3;
  const float c0 = Bc[n * 3 + 0];
  const float c1 = Bc[n * 3 + 1];
  const float c2 = Bc[n * 3 + 2];
  const float4 a = ((const float4*)(X + ((size_t)nm * 3 + 0) * DD))[d4];
  const float4 b = ((const float4*)(X + ((size_t)nm * 3 + 1) * DD))[d4];
  const float4 c = ((const float4*)(X + ((size_t)nm * 3 + 2) * DD))[d4];
  float4 r;
  r.x = c0 * a.x + c1 * b.x + c2 * c.x;
  r.y = c0 * a.y + c1 * b.y + c2 * c.y;
  r.z = c0 * a.z + c1 * b.z + c2 * c.z;
  r.w = c0 * a.w + c1 * b.w + c2 * c.w;
  ((float4*)(y + (size_t)nm * DD))[d4] = r;
}

// ---------------- host launcher ----------------
extern "C" void kernel_launch(void* const* d_in, const int* in_sizes, int n_in,
                              void* d_out, int out_size, void* d_ws, size_t ws_size,
                              hipStream_t stream) {
  const float* E  = (const float*)d_in[0];
  const float* X  = (const float*)d_in[1];
  const float* W1 = (const float*)d_in[2];
  const float* b1 = (const float*)d_in[3];
  const float* W2 = (const float*)d_in[4];
  const float* b2 = (const float*)d_in[5];
  const float* W3 = (const float*)d_in[6];
  const float* b3 = (const float*)d_in[7];
  float* y = (float*)d_out;

  char* ws = (char*)d_ws;
  const size_t MB = 1024 * 1024;
  __bf16* h1  = (__bf16*)(ws);             //  8 MB: [4096,1024] bf16
  __bf16* h2  = (__bf16*)(ws + 8  * MB);   //  8 MB
  float*  Bc  = (float*) (ws + 16 * MB);   // 48 KB: [4096,3] f32
  __bf16* Ebf = (__bf16*)(ws + 17 * MB);   // 16 MB: [4096,2048] bf16
  __bf16* W1t = (__bf16*)(ws + 33 * MB);   //  4 MB: [1024,2048] bf16 (transposed)
  __bf16* W2t = (__bf16*)(ws + 37 * MB);   //  2 MB: [1024,1024] bf16 (transposed)

  const dim3 blk(256);
  // one-time bf16 conversion / weight transpose (K-contiguous operands)
  cvt_bf16_kernel<<<(NN * NEMB / 4) / 256, blk, 0, stream>>>(E, Ebf);
  transpose_cvt_kernel<<<(NHID * (NEMB / 16)) / 256, blk, 0, stream>>>(W1, W1t, NEMB);
  transpose_cvt_kernel<<<(NHID * (NHID / 16)) / 256, blk, 0, stream>>>(W2, W2t, NHID);
  // h1 = elu(E @ W1 + b1)   [4096, 1024]
  gemm_bias_elu_wmma<<<dim3(NHID / 128, NN / 128), blk, 0, stream>>>(
      Ebf, NEMB, W1t, b1, h1, NHID, NEMB);
  // h2 = elu(h1 @ W2 + b2)  [4096, 1024]
  gemm_bias_elu_wmma<<<dim3(NHID / 128, NN / 128), blk, 0, stream>>>(
      h1, NHID, W2t, b2, h2, NHID, NHID);
  // B = h2 @ W3 + b3        [4096, 3]
  coef_kernel<<<NN / 8, blk, 0, stream>>>(h2, W3, b3, Bc);
  // y = einsum("nc,nmcd->nmd", B, X)
  apply_kernel<<<(NN * MM * DD / 4) / 256, blk, 0, stream>>>(Bc, X, y);
}
